// TopKRouter_74964359184846
// MI455X (gfx1250) — matrix operand order repaired and checked by reading, not compiled, not run on hardware
//
#include <hip/hip_runtime.h>
#include <hip/hip_bf16.h>

typedef __attribute__((ext_vector_type(16))) __bf16 v16bf;
typedef __attribute__((ext_vector_type(8)))  float  v8f;
typedef __attribute__((ext_vector_type(4)))  float  f32x4;
typedef __attribute__((ext_vector_type(4)))  unsigned int u32x4;
typedef __attribute__((ext_vector_type(8)))  int i32x8;
typedef __attribute__((ext_vector_type(4)))  int i32x4;

#define DIM        4096
#define NEXP       64
#define TOK_PER_WG 128
#define KSLAB      256
#define KSTEP      32
#define B_STRIDE   264   // 256 + 8 bf16 pad -> conflict-free ds_load_b128 across 16 experts
#define L_STRIDE   65    // 64 + 1 pad      -> conflict-free scalar top-2 scan

// ---------------- gate_w f32 -> bf16 (one-time, lives in L2) ----------------
__global__ void gate_to_bf16(const float* __restrict__ g, __bf16* __restrict__ gb, int n) {
    int i = blockIdx.x * blockDim.x + threadIdx.x;
    if (i < n) gb[i] = (__bf16)g[i];
}

// ---------------- main router: TDM-staged bf16 WMMA GEMM + top-2 epilogue ----------------
__global__ __launch_bounds__(256)
void router_kernel(const float* __restrict__ x,
                   const __bf16* __restrict__ gb,
                   float* __restrict__ wout,
                   int* __restrict__ iout)
{
    __shared__ __bf16 sB[2][NEXP * B_STRIDE];    // 2 x ~33 KB gate slabs (double buffer)
    __shared__ float  sL[TOK_PER_WG * L_STRIDE]; // ~33 KB logits

    const int tid   = threadIdx.x;
    const int lane  = tid & 31;
    const int wave  = tid >> 5;       // 8 waves
    const int lhalf = lane >> 4;      // 0: lanes 0-15, 1: lanes 16-31
    const int l16   = lane & 15;

    const int tok0 = blockIdx.x * TOK_PER_WG;
    const int row  = tok0 + wave * 16 + l16;     // A-matrix row for this lane
    const float* xrow = x + (size_t)row * DIM;

    v8f acc0 = {}, acc1 = {}, acc2 = {}, acc3 = {};

    // TDM descriptor issue: pull [64 experts x 256 K] bf16 tile into LDS buffer `buf`,
    // with hardware padding of 4 DWORDs after every 128 DWORDs (=> 264-element rows).
    auto issue_tdm = [&](int ko, int buf) {
        u32x4 g0;
        g0.x = 1u;                                              // count=1 valid descriptor
        g0.y = (unsigned int)(size_t)(&sB[buf][0]);             // LDS byte address (low 32 bits)
        unsigned long long ga = (unsigned long long)(size_t)(gb + (size_t)ko * KSLAB);
        g0.z = (unsigned int)ga;                                // global_addr[31:0]
        g0.w = (unsigned int)(ga >> 32) | (2u << 30);           // global_addr[56:32] | type=2

        i32x8 g1;
        g1[0] = (1 << 16)   // data_size = 1 -> 2 bytes (bf16)
              | (1 << 20)   // pad_enable
              | (6 << 22)   // pad_interval: every 128 DWORDs (one 256-bf16 row)
              | (3 << 25);  // pad_amount: 4 DWORDs (8 bf16) -> 264-element row stride
        g1[1] = (DIM & 0xFFFF) << 16;                 // tensor_dim0[15:0] @ bits 63:48
        g1[2] = (DIM >> 16) | (NEXP << 16);           // tensor_dim0[31:16] | tensor_dim1[15:0]
        g1[3] = (KSLAB << 16);                        // tensor_dim1[31:16]=0 | tile_dim0=256
        g1[4] = NEXP;                                 // tile_dim1=64, tile_dim2=0
        g1[5] = DIM;                                  // tensor_dim0_stride[31:0] = 4096 elems
        g1[6] = 0;                                    // stride hi | tensor_dim1_stride lo
        g1[7] = 0;
        i32x4 gz4 = {0, 0, 0, 0};                     // groups 2/3: 2D tensor, unused dims
        i32x8 gz8 = {0, 0, 0, 0, 0, 0, 0, 0};
        __builtin_amdgcn_tensor_load_to_lds(g0, g1, gz4, gz4, gz8, 0);
    };

    if (wave == 0) issue_tdm(0, 0);                   // prime the pipeline

    for (int ko = 0; ko < DIM / KSLAB; ++ko) {
        const int buf = ko & 1;
        if (wave == 0) {
            if (ko + 1 < DIM / KSLAB) {
                issue_tdm(ko + 1, buf ^ 1);           // prefetch next slab
                __builtin_amdgcn_s_wait_tensorcnt(1); // in-order: slab `ko` is resident
            } else {
                __builtin_amdgcn_s_wait_tensorcnt(0);
            }
        }
        __syncthreads();                              // release slab `ko` to all waves
        const __bf16* Bs = &sB[buf][0];

        #pragma unroll
        for (int ks = 0; ks < KSLAB / KSTEP; ++ks) {
            const int k = ko * KSLAB + ks * KSTEP;

            // ---- A fragment (16x32 bf16), converted f32->bf16 in registers ----
            // lane<16 holds K {0..7, 16..23}; lane>=16 holds K {8..15, 24..31}
            union { v16bf v; __bf16 h[16]; } a;
            {
                const int ka = k + lhalf * 8;
                f32x4 c0 = *(const f32x4*)(xrow + ka);
                f32x4 c1 = *(const f32x4*)(xrow + ka + 4);
                f32x4 c2 = *(const f32x4*)(xrow + ka + 16);
                f32x4 c3 = *(const f32x4*)(xrow + ka + 20);
                #pragma unroll
                for (int j = 0; j < 4; ++j) {
                    a.h[j]      = (__bf16)c0[j];
                    a.h[4 + j]  = (__bf16)c1[j];
                    a.h[8 + j]  = (__bf16)c2[j];
                    a.h[12 + j] = (__bf16)c3[j];
                }
            }

            // ---- B fragments from LDS: expert column = lane&15, K {0..7,16..23} / {8..15,24..31} ----
            const int kk = ks * KSTEP + lhalf * 8;
            union { v16bf v; uint4 q[2]; } b0, b1, b2, b3;
            b0.q[0] = *(const uint4*)(&Bs[( 0 + l16) * B_STRIDE + kk]);
            b0.q[1] = *(const uint4*)(&Bs[( 0 + l16) * B_STRIDE + kk + 16]);
            b1.q[0] = *(const uint4*)(&Bs[(16 + l16) * B_STRIDE + kk]);
            b1.q[1] = *(const uint4*)(&Bs[(16 + l16) * B_STRIDE + kk + 16]);
            b2.q[0] = *(const uint4*)(&Bs[(32 + l16) * B_STRIDE + kk]);
            b2.q[1] = *(const uint4*)(&Bs[(32 + l16) * B_STRIDE + kk + 16]);
            b3.q[0] = *(const uint4*)(&Bs[(48 + l16) * B_STRIDE + kk]);
            b3.q[1] = *(const uint4*)(&Bs[(48 + l16) * B_STRIDE + kk + 16]);

            acc0 = __builtin_amdgcn_wmma_f32_16x16x32_bf16(false, a.v, false, b0.v, (short)0, acc0, false, false);
            acc1 = __builtin_amdgcn_wmma_f32_16x16x32_bf16(false, a.v, false, b1.v, (short)0, acc1, false, false);
            acc2 = __builtin_amdgcn_wmma_f32_16x16x32_bf16(false, a.v, false, b2.v, (short)0, acc2, false, false);
            acc3 = __builtin_amdgcn_wmma_f32_16x16x32_bf16(false, a.v, false, b3.v, (short)0, acc3, false, false);
        }
        __syncthreads();                              // all reads of this slab done
    }

    // ---- scatter logits to LDS (C layout: VGPR r -> M=r+8*lhalf, N=lane&15) ----
    #pragma unroll
    for (int r = 0; r < 8; ++r) {
        int tl = wave * 16 + r + lhalf * 8;           // token within workgroup
        sL[tl * L_STRIDE +  0 + l16] = acc0[r];
        sL[tl * L_STRIDE + 16 + l16] = acc1[r];
        sL[tl * L_STRIDE + 32 + l16] = acc2[r];
        sL[tl * L_STRIDE + 48 + l16] = acc3[r];
    }
    __syncthreads();

    // ---- top-2 + renormalized softmax (only top-2 logits matter) ----
    if (tid < TOK_PER_WG) {
        const float* lrow = &sL[tid * L_STRIDE];
        float best = -3.402823466e38f, sec = -3.402823466e38f;
        int bi = 0, si = 0;
        for (int e = 0; e < NEXP; ++e) {
            float v = lrow[e];
            if (v > best)      { sec = best; si = bi; best = v; bi = e; }
            else if (v > sec)  { sec = v; si = e; }
        }
        float e2 = __expf(sec - best);                // <= 1
        float w0 = 1.0f / (1.0f + e2);
        float w1 = 1.0f - w0;
        int g = tok0 + tid;
        wout[2 * g + 0] = w0;
        wout[2 * g + 1] = w1;
        iout[2 * g + 0] = bi;
        iout[2 * g + 1] = si;
    }
}

extern "C" void kernel_launch(void* const* d_in, const int* in_sizes, int n_in,
                              void* d_out, int out_size, void* d_ws, size_t ws_size,
                              hipStream_t stream) {
    const float* x  = (const float*)d_in[0];   // [8,4096,4096] f32
    const float* gw = (const float*)d_in[1];   // [64,4096] f32
    const int gw_n  = in_sizes[1];             // 262144
    const int ntok  = in_sizes[0] / DIM;       // 32768

    __bf16* gb = (__bf16*)d_ws;                // 512 KB bf16 gate in scratch

    float* wout = (float*)d_out;               // first half: top-2 weights
    int*   iout = (int*)((float*)d_out + out_size / 2); // second half: indices

    gate_to_bf16<<<(gw_n + 255) / 256, 256, 0, stream>>>(gw, gb, gw_n);
    router_kernel<<<ntok / TOK_PER_WG, 256, 0, stream>>>(x, gb, wout, iout);
}
